// SelfAttention_21887153340534
// MI455X (gfx1250) — compile-verified, matrix operand order checked by reading, not run on hardware
//
#include <hip/hip_runtime.h>
#include <hip/hip_bf16.h>

typedef __attribute__((ext_vector_type(16))) __bf16 v16bf;
typedef __attribute__((ext_vector_type(8)))  float  v8f;
typedef unsigned int u32;
typedef __attribute__((ext_vector_type(4))) u32 v4u;
typedef __attribute__((ext_vector_type(8))) int  v8i;
typedef __attribute__((ext_vector_type(4))) int  v4i;

constexpr int BATCH = 8;
constexpr int SEQ   = 2048;
constexpr int DIN   = 1024;
constexpr int DKN   = 1024;
constexpr int MTOT  = BATCH * SEQ;   // 16384 total query rows

// ---------------------------------------------------------------------------
// Fragment loader per CDNA5 ISA 7.12.2 (wave32):
//  16-bit A 16x32 (MxK): lane L<16 -> row L,  K in {0..7, 16..23}
//                        lane L>=16 -> row L-16, K in {8..15, 24..31}
//  (also used for B when the tile is stored K-contiguous per column)
//  C/D 16x16 f32: VGPR j -> M = j + 8*(lane>=16), N = lane & 15
// ---------------------------------------------------------------------------
__device__ __forceinline__ v16bf load_frag_row(const __bf16* base, int ld) {
  const int lane = threadIdx.x & 31;
  const int r = lane & 15;
  const int h = lane >> 4;
  const __bf16* p = base + (size_t)r * ld + h * 8;
  v16bf f;
#pragma unroll
  for (int i = 0; i < 8; ++i) f[i] = p[i];
#pragma unroll
  for (int i = 0; i < 8; ++i) f[8 + i] = p[16 + i];
  return f;
}

__device__ __forceinline__ v8f wmma_bf16(v16bf a, v16bf b, v8f c) {
  return __builtin_amdgcn_wmma_f32_16x16x32_bf16(
      /*neg_a=*/false, a, /*neg_b=*/false, b,
      /*c_mod=*/(short)0, c, /*reuse_a=*/false, /*reuse_b=*/false);
}

__device__ __forceinline__ unsigned pack_bf16(float a, float b) {
  unsigned short lo = __builtin_bit_cast(unsigned short, (__bf16)a);
  unsigned short hi = __builtin_bit_cast(unsigned short, (__bf16)b);
  return (unsigned)lo | ((unsigned)hi << 16);
}

// ---------------------------------------------------------------------------
// TDM: issue a 2-D TENSOR_LOAD_TO_LDS (bf16 elements) via hand-built D#.
// D# field packing per CDNA5 ISA ch.8 (group0 128b, group1 256b).
// tile: tile_d0 x tile_d1 elements, line stride stride_d0 elements; rows land
// densely packed in LDS (tile_d0 elements per line).
// 6-arg builtin (clang-23 lane): (v4u g0, v8i g1, v4i g2, v4i g3, v8i g4, cpol)
// ---------------------------------------------------------------------------
__device__ __forceinline__ void tdm_load_2d_bf16(u32 lds_off, const void* gptr,
                                                 u32 tile_d0, u32 tile_d1,
                                                 u32 tensor_d0, u32 tensor_d1,
                                                 u32 stride_d0) {
  unsigned long long ga = (unsigned long long)gptr;
  v4u g0;
  g0[0] = 1u;                                       // count=1, user descriptor
  g0[1] = lds_off;                                  // lds_addr (bytes)
  g0[2] = (u32)(ga & 0xFFFFFFFFu);                  // global_addr[31:0]
  g0[3] = (u32)((ga >> 32) & 0x1FFFFFFu) | (2u << 30);  // addr[56:32] | type=2
  v8i g1;
  g1[0] = (int)(1u << 16);                          // wg_mask=0, data_size=1 (2B)
  g1[1] = (int)((tensor_d0 & 0xFFFFu) << 16);       // tensor_dim0[15:0]
  g1[2] = (int)((tensor_d0 >> 16) | ((tensor_d1 & 0xFFFFu) << 16));
  g1[3] = (int)((tensor_d1 >> 16) | (tile_d0 << 16));   // tile_dim0
  g1[4] = (int)(tile_d1 & 0xFFFFu);                 // tile_dim1 (tile_dim2=0)
  g1[5] = (int)stride_d0;                           // tensor_dim0_stride[31:0]
  g1[6] = 0;                                        // stride0 hi | stride1 lo
  g1[7] = 0;
  v4i z4 = {0, 0, 0, 0};                            // 2-D: groups 2/3 unused
  v8i z8 = {0, 0, 0, 0, 0, 0, 0, 0};
  __builtin_amdgcn_tensor_load_to_lds(g0, g1, z4, z4, z8, 0);
}

// ---------------------------------------------------------------------------
// Kernel 1: out = x * W + bias  (bf16 output)
//   TRANSPOSE_OUT=false: out[b*SEQ+row][col]   ([MTOT][DKN], for Q and K)
//   TRANSPOSE_OUT=true : out[b][col][row]      ([BATCH][DKN][SEQ], for V^T)
// Block: 256 threads (8 waves), tile 128x64, K-step 32.
// ---------------------------------------------------------------------------
template <bool TRANSPOSE_OUT>
__global__ __launch_bounds__(256)
void qkv_proj_kernel(const float* __restrict__ x, const float* __restrict__ W,
                     const float* __restrict__ bias, __bf16* __restrict__ out) {
  __shared__ __bf16 As[128 * 32];   // x tile, row-major [m][k]   (8 KB)
  __shared__ __bf16 Bt[64 * 32];    // W tile, transposed [n][k]  (4 KB)
  const int m0 = blockIdx.x * 128;
  const int n0 = blockIdx.y * 64;
  const int tid = threadIdx.x;
  const int wid = tid >> 5;
  const int lane = tid & 31;
  const int wm = wid >> 1;          // 4 wave rows x 2 wave cols (32x32 each)
  const int wn = wid & 1;

  v8f acc[2][2] = {};
  for (int k0 = 0; k0 < DIN; k0 += 32) {
    // stage x tile (128x32): float4 loads, packed bf16x4 b64 stores
#pragma unroll
    for (int i = 0; i < 4; ++i) {
      int idx = tid + i * 256;            // float4 id 0..1023
      int r = idx >> 3, c4 = idx & 7;     // 8 float4 per 32-wide row
      const float4 f =
          ((const float4*)(x + (size_t)(m0 + r) * DIN + k0))[c4];
      uint2 p;
      p.x = pack_bf16(f.x, f.y);
      p.y = pack_bf16(f.z, f.w);
      ((uint2*)As)[r * 8 + c4] = p;
    }
    // stage W tile (32x64) transposed: float4 loads, 4 scattered b16 stores
#pragma unroll
    for (int i = 0; i < 2; ++i) {
      int idx = tid + i * 256;            // float4 id 0..511
      int kk = idx >> 4, n4 = idx & 15;   // 16 float4 per 64-wide row
      const float4 f =
          ((const float4*)(W + (size_t)(k0 + kk) * DKN + n0))[n4];
      Bt[(n4 * 4 + 0) * 32 + kk] = (__bf16)f.x;
      Bt[(n4 * 4 + 1) * 32 + kk] = (__bf16)f.y;
      Bt[(n4 * 4 + 2) * 32 + kk] = (__bf16)f.z;
      Bt[(n4 * 4 + 3) * 32 + kk] = (__bf16)f.w;
    }
    __syncthreads();
    v16bf a0 = load_frag_row(As + (wm * 32) * 32, 32);
    v16bf a1 = load_frag_row(As + (wm * 32 + 16) * 32, 32);
    v16bf b0 = load_frag_row(Bt + (wn * 32) * 32, 32);
    v16bf b1 = load_frag_row(Bt + (wn * 32 + 16) * 32, 32);
    acc[0][0] = wmma_bf16(a0, b0, acc[0][0]);
    acc[0][1] = wmma_bf16(a0, b1, acc[0][1]);
    acc[1][0] = wmma_bf16(a1, b0, acc[1][0]);
    acc[1][1] = wmma_bf16(a1, b1, acc[1][1]);
    __syncthreads();
  }
  // epilogue: + bias, store bf16
  const int h = lane >> 4;
  const int nn = lane & 15;
#pragma unroll
  for (int tm = 0; tm < 2; ++tm) {
#pragma unroll
    for (int tn = 0; tn < 2; ++tn) {
      const int col = n0 + wn * 32 + tn * 16 + nn;
      const float bvv = bias[col];
      if constexpr (!TRANSPOSE_OUT) {
#pragma unroll
        for (int j = 0; j < 8; ++j) {
          const int row = m0 + wm * 32 + tm * 16 + j + 8 * h;
          out[(size_t)row * DKN + col] = (__bf16)(acc[tm][tn][j] + bvv);
        }
      } else {
        // V^T: the 8 rows per lane are consecutive seq positions -> one b128
        const int row0 = m0 + wm * 32 + tm * 16 + 8 * h;
        const int b = row0 >> 11;          // row / SEQ
        const int sr = row0 & (SEQ - 1);
        uint4 pk;
        pk.x = pack_bf16(acc[tm][tn][0] + bvv, acc[tm][tn][1] + bvv);
        pk.y = pack_bf16(acc[tm][tn][2] + bvv, acc[tm][tn][3] + bvv);
        pk.z = pack_bf16(acc[tm][tn][4] + bvv, acc[tm][tn][5] + bvv);
        pk.w = pack_bf16(acc[tm][tn][6] + bvv, acc[tm][tn][7] + bvv);
        *(uint4*)(out + ((size_t)b * DKN + col) * SEQ + sr) = pk;
      }
    }
  }
}

// ---------------------------------------------------------------------------
// Kernel 2: fused attention, one batch / 16 query rows per block.
// LDS (dynamic, ~230 KB of the 320 KB WGP pool):
//   Qs  : 16x1024 bf16  ( 32 KB)
//   Ss  : 16x2048 f32   (128 KB)  -- reused as double-buffered V^T stage
//   Ws  : 16x2048 bf16  ( 64 KB)
//   red : 16x17 f32
// V^T tiles (1024 cols x 32 seq, stride SEQ) are streamed by the Tensor Data
// Mover into LDS as [1024][32] row-major, so B fragments are two ds_load_b128.
// ---------------------------------------------------------------------------
__global__ __launch_bounds__(256)
void attn_kernel(const __bf16* __restrict__ qb, const __bf16* __restrict__ kb,
                 const __bf16* __restrict__ vtb, float* __restrict__ outp) {
  extern __shared__ char smem[];
  __bf16* Qs = (__bf16*)smem;
  float*  Ss = (float*)(smem + 32768);
  __bf16* Ws = (__bf16*)(smem + 32768 + 131072);
  float*  red = (float*)(smem + 32768 + 131072 + 65536);
  __bf16* Vt0 = (__bf16*)Ss;                  // [1024][32] bf16 (64 KB)
  __bf16* Vt1 = Vt0 + DKN * 32;               // second buffer (64 KB)

  const int qt  = blockIdx.x;        // query tile (16 rows)
  const int bb  = blockIdx.y;        // batch
  const int m0  = qt * 16;
  const int tid = threadIdx.x;
  const int wid = tid >> 5;
  const int lane = tid & 31;
  const int h = lane >> 4;
  const int nn = lane & 15;

  const __bf16* qg = qb + ((size_t)bb * SEQ + m0) * DKN;
  const __bf16* kg = kb + (size_t)bb * SEQ * DKN;
  const __bf16* vtg = vtb + (size_t)bb * DKN * SEQ;   // V^T[col][seq]

  // --- Phase 1: Q tile -> LDS (16x1024 bf16 = 2048 uint4, 8/thread) ---
  {
    const uint4* src = (const uint4*)qg;
    uint4* dst = (uint4*)Qs;
#pragma unroll
    for (int i = 0; i < 8; ++i) dst[tid + i * 256] = src[tid + i * 256];
  }
  __syncthreads();

  // --- Phase 2: S = (Q K^T) * scale, 16x2048 into LDS ---
  const float scale = 0.03125f;  // 1/sqrt(1024)
  for (int kt = wid; kt < SEQ / 16; kt += 8) {
    const __bf16* kbase = kg + (size_t)kt * 16 * DKN;
    if (kt + 8 < SEQ / 16)                       // prefetch next key tile
      __builtin_prefetch(kg + (size_t)(kt + 8) * 16 * DKN, 0, 0);
    v8f acc = {};
    for (int kk = 0; kk < DKN; kk += 32) {
      v16bf a = load_frag_row(Qs + kk, DKN);
      v16bf b = load_frag_row(kbase + kk, DKN);
      acc = wmma_bf16(a, b, acc);
    }
#pragma unroll
    for (int j = 0; j < 8; ++j)
      Ss[(size_t)(j + 8 * h) * SEQ + kt * 16 + nn] = acc[j] * scale;
  }
  __syncthreads();

  // --- Phase 3: row-wise softmax in LDS; bf16 weights into Ws ---
  {
    const int r = tid >> 4;
    const int c = tid & 15;
    float* srow = Ss + (size_t)r * SEQ;
    float lmax = -3.4e38f;
    for (int i = c; i < SEQ; i += 16) lmax = fmaxf(lmax, srow[i]);
    red[r * 17 + c] = lmax;
    __syncthreads();
    if (c == 0) {
      float m = red[r * 17];
#pragma unroll
      for (int i = 1; i < 16; ++i) m = fmaxf(m, red[r * 17 + i]);
      red[r * 17 + 16] = m;
    }
    __syncthreads();
    const float rmax = red[r * 17 + 16];
    float lsum = 0.f;
    for (int i = c; i < SEQ; i += 16) {
      float e = __expf(srow[i] - rmax);
      srow[i] = e;
      lsum += e;
    }
    red[r * 17 + c] = lsum;
    __syncthreads();
    if (c == 0) {
      float s = red[r * 17];
#pragma unroll
      for (int i = 1; i < 16; ++i) s += red[r * 17 + i];
      red[r * 17 + 16] = s;
    }
    __syncthreads();
    const float inv = 1.0f / red[r * 17 + 16];
    __bf16* wrow = Ws + (size_t)r * SEQ;
    for (int i = c; i < SEQ; i += 16) wrow[i] = (__bf16)(srow[i] * inv);
  }
  __syncthreads();   // Ss (score region) now dead -> reusable as V stage

  // --- Phase 4: O = P * V, V^T streamed by TDM (double-buffered) ---
  // TDM tile: d0 = 32 seq (64 B lines), d1 = 1024 cols, line stride SEQ.
  const u32 vt_off0 = (u32)(uintptr_t)Vt0;    // LDS byte offsets for D#
  const u32 vt_off1 = (u32)(uintptr_t)Vt1;

  if (wid == 0)   // prologue: tile 0 -> buffer 0
    tdm_load_2d_bf16(vt_off0, vtg, /*tile*/ 32, DKN,
                     /*tensor*/ SEQ, DKN, /*stride*/ SEQ);

  v8f oacc[8] = {};
  const int ncol0 = wid * 128;
  constexpr int NTILES = SEQ / 32;            // 64
  for (int it = 0; it < NTILES; ++it) {
    const int s0 = it * 32;
    if (wid == 0) {
      if (it + 1 < NTILES) {                  // issue next tile into other buf
        tdm_load_2d_bf16((it & 1) ? vt_off0 : vt_off1,
                         vtg + (size_t)(s0 + 32),
                         32, DKN, SEQ, DKN, SEQ);
        __builtin_amdgcn_s_wait_tensorcnt(1); // oldest (current tile) done
      } else {
        __builtin_amdgcn_s_wait_tensorcnt(0);
      }
    }
    __syncthreads();                          // publish TDM data to all waves
    const __bf16* Vcur = (it & 1) ? Vt1 : Vt0;
    v16bf a = load_frag_row(Ws + s0, SEQ);    // one A fragment, reused 8x
#pragma unroll
    for (int t = 0; t < 8; ++t) {
      // B fragment: lane n -> LDS row (ncol0 + t*16 + n), contiguous along s
      v16bf b = load_frag_row(Vcur + (size_t)(ncol0 + t * 16) * 32, 32);
      oacc[t] = wmma_bf16(a, b, oacc[t]);
    }
    __syncthreads();                          // done reading before overwrite
  }

  // store fp32 output
  float* og = outp + ((size_t)bb * SEQ + m0) * DKN;
#pragma unroll
  for (int t = 0; t < 8; ++t) {
    const int col = ncol0 + t * 16 + nn;
#pragma unroll
    for (int j = 0; j < 8; ++j)
      og[(size_t)(j + 8 * h) * DKN + col] = oacc[t][j];
  }
}

// ---------------------------------------------------------------------------
extern "C" void kernel_launch(void* const* d_in, const int* in_sizes, int n_in,
                              void* d_out, int out_size, void* d_ws, size_t ws_size,
                              hipStream_t stream) {
  const float* x  = (const float*)d_in[0];
  const float* Wq = (const float*)d_in[1];
  const float* bq = (const float*)d_in[2];
  const float* Wk = (const float*)d_in[3];
  const float* bk = (const float*)d_in[4];
  const float* Wv = (const float*)d_in[5];
  const float* bv = (const float*)d_in[6];
  float* out = (float*)d_out;

  // workspace: Q,K row-major + V transposed, bf16, 32 MB each (96 MB total)
  __bf16* qbuf = (__bf16*)d_ws;
  __bf16* kbuf = qbuf + (size_t)MTOT * DKN;
  __bf16* vtbuf = kbuf + (size_t)MTOT * DKN;   // [BATCH][DKN][SEQ]

  dim3 gproj(MTOT / 128, DKN / 64);   // 128 x 16 blocks
  qkv_proj_kernel<false><<<gproj, 256, 0, stream>>>(x, Wq, bq, qbuf);
  qkv_proj_kernel<false><<<gproj, 256, 0, stream>>>(x, Wk, bk, kbuf);
  qkv_proj_kernel<true ><<<gproj, 256, 0, stream>>>(x, Wv, bv, vtbuf);

  const size_t smem = 32768 + 131072 + 65536 + (size_t)16 * 17 * 4;  // ~230 KB
  attn_kernel<<<dim3(SEQ / 16, BATCH), 256, smem, stream>>>(qbuf, kbuf, vtbuf, out);
}
